// GE2ELoss_90787018703151
// MI455X (gfx1250) — matrix operand order repaired and checked by reading, not compiled, not run on hardware
//
#include <hip/hip_runtime.h>
#include <hip/hip_bf16.h>

// GE2E loss for S=2048 speakers, U=10 utterances, D=256 dims (fp32 in, fp32 scalar out).
// Pipeline:
//   1) prep_rows:  normalize each of 20480 embedding rows -> f16 En_h   (ws)
//   2) prep_cent:  mean over U + normalize -> f16 Cn_h (2048x256)       (ws)
//   3) ge2e_sim:   async-LDS double-buffered B tiles (global_load_async_to_lds_b128,
//                  ASYNCcnt) + WMMA f32_16x16x32_f16, fused streaming logsumexp
//                  (fixed shift of 1.0 since |sim|<=1) -> per-row loss
//   4) final_red:  deterministic single-block mean -> d_out[0]

#define S_SPK 2048
#define U_UTT 10
#define D_DIM 256
#define B_ROWS (S_SPK * U_UTT)   // 20480
#define GEPS 1e-8f

typedef __attribute__((ext_vector_type(16))) _Float16 v16h;
typedef __attribute__((ext_vector_type(8)))  _Float16 v8h;
typedef __attribute__((ext_vector_type(8)))  float    v8f;

// ---------------- phase 1a: normalize embedding rows to f16 ----------------
__global__ void __launch_bounds__(256) prep_rows(const float* __restrict__ E,
                                                 _Float16* __restrict__ EnH) {
  const int wave = (blockIdx.x * blockDim.x + threadIdx.x) >> 5;  // one wave per row
  const int lane = threadIdx.x & 31;
  const float* row = E + (size_t)wave * D_DIM + lane * 8;
  float v[8];
  float ss = 0.f;
#pragma unroll
  for (int i = 0; i < 8; ++i) { v[i] = row[i]; ss += v[i] * v[i]; }
#pragma unroll
  for (int m = 16; m >= 1; m >>= 1) ss += __shfl_xor(ss, m, 32);
  const float inv = 1.0f / fmaxf(sqrtf(ss), GEPS);
  _Float16* out = EnH + (size_t)wave * D_DIM + lane * 8;
#pragma unroll
  for (int i = 0; i < 8; ++i) out[i] = (_Float16)(v[i] * inv);
}

// ---------------- phase 1b: centroids (mean over U) + normalize -> f16 -----
__global__ void __launch_bounds__(256) prep_cent(const float* __restrict__ E,
                                                 _Float16* __restrict__ CnH) {
  const int wave = (blockIdx.x * blockDim.x + threadIdx.x) >> 5;  // one wave per speaker
  const int lane = threadIdx.x & 31;
  const float* base = E + (size_t)wave * U_UTT * D_DIM + lane * 8;
  float v[8];
#pragma unroll
  for (int i = 0; i < 8; ++i) v[i] = 0.f;
  for (int u = 0; u < U_UTT; ++u) {
#pragma unroll
    for (int i = 0; i < 8; ++i) v[i] += base[(size_t)u * D_DIM + i];
  }
  float ss = 0.f;
#pragma unroll
  for (int i = 0; i < 8; ++i) { v[i] *= (1.0f / U_UTT); ss += v[i] * v[i]; }
#pragma unroll
  for (int m = 16; m >= 1; m >>= 1) ss += __shfl_xor(ss, m, 32);
  const float inv = 1.0f / fmaxf(sqrtf(ss), GEPS);
  _Float16* out = CnH + (size_t)wave * D_DIM + lane * 8;
#pragma unroll
  for (int i = 0; i < 8; ++i) out[i] = (_Float16)(v[i] * inv);
}

// Async copy of one 16x256-half (8 KB) centroid tile into LDS.
// 32 lanes x 16 B x 16 instructions = 8 KB, tracked on the issuing wave's ASYNCcnt.
__device__ __forceinline__ void issue_b_tile_async(const _Float16* __restrict__ gsrc,
                                                   _Float16* __restrict__ ldst,
                                                   int lane) {
  const char* g = (const char*)gsrc + lane * 16;
  unsigned int l = (unsigned int)(uintptr_t)ldst + (unsigned int)(lane * 16);
#pragma unroll
  for (int i = 0; i < 16; ++i) {
    unsigned int laddr = l + i * 512;
    const char* gaddr = g + (size_t)i * 512;
    asm volatile("global_load_async_to_lds_b128 %0, %1, off"
                 :: "v"(laddr), "v"(gaddr)
                 : "memory");
  }
}

__device__ __forceinline__ void wait_async_all() {
#if __has_builtin(__builtin_amdgcn_s_wait_asynccnt)
  __builtin_amdgcn_s_wait_asynccnt(0);
#else
  asm volatile("s_wait_asynccnt 0x0" ::: "memory");
#endif
}

// ---------------- phase 2: WMMA GEMM + fused streaming LSE -----------------
// One wave owns a 16-row M-tile of En_h (full K=256 A panel in 64 VGPRs).
// The 8 waves of a block share double-buffered B tiles staged in LDS by the
// async copy unit: copy of tile nt+1 overlaps compute of tile nt.
__global__ void __launch_bounds__(256) ge2e_sim(const _Float16* __restrict__ EnH,
                                                const _Float16* __restrict__ CnH,
                                                float* __restrict__ loss) {
  __shared__ _Float16 bsh[2][16 * D_DIM];  // 2 x 8 KB double buffer

  const int tid  = threadIdx.x;
  const int widx = tid >> 5;
  const int lane = tid & 31;
  const int wave = blockIdx.x * 8 + widx;  // 0..1279
  const int lm   = lane & 15;              // A: row-in-tile; B: col-in-tile; C: N
  const int hi   = lane >> 4;              // half-wave select
  const int rowbase = wave * 16;

  // A panel: 16x32 f16 layout -> lane m=lm, K chunk kb..kb+15, kb = hi?16:0
  v16h A[8];
  {
    const _Float16* arow = EnH + (size_t)(rowbase + lm) * D_DIM + (hi ? 16 : 0);
#pragma unroll
    for (int ks = 0; ks < 8; ++ks) {
      v8h lo = *(const v8h*)(arow + ks * 32);
      v8h hh = *(const v8h*)(arow + ks * 32 + 8);
      A[ks] = __builtin_shufflevector(lo, hh, 0, 1, 2, 3, 4, 5, 6, 7,
                                      8, 9, 10, 11, 12, 13, 14, 15);
    }
  }

  float acc[8], pos[8];
#pragma unroll
  for (int j = 0; j < 8; ++j) { acc[j] = 0.f; pos[j] = 0.f; }

  // Prologue: stage tile 0.
  if (widx == 0) {
    issue_b_tile_async(CnH, &bsh[0][0], lane);
    wait_async_all();
  }
  __syncthreads();

  for (int nt = 0; nt < S_SPK / 16; ++nt) {
    const int cur = nt & 1;

    // Prefetch tile nt+1 into the other buffer (safe: its last readers finished
    // at the end-of-iteration barrier of nt-1).
    if (widx == 0 && nt + 1 < S_SPK / 16) {
      issue_b_tile_async(CnH + (size_t)(nt + 1) * 16 * D_DIM, &bsh[cur ^ 1][0], lane);
    }

    // B: 32x16 f16, B[k][n] = Cn[n][k]; lane n=lm, K chunk kb..kb+15, kb = hi?16:0
    // Hoist all 8 chunks (16x ds_load_b128) so the scheduler can clause the LDS
    // loads and issue the 8 WMMAs back-to-back with partial dscnt waits.
    const _Float16* brow = &bsh[cur][(size_t)lm * D_DIM + (hi ? 16 : 0)];
    v16h B[8];
#pragma unroll
    for (int ks = 0; ks < 8; ++ks) B[ks] = *(const v16h*)(brow + ks * 32);

    v8f c = {};
#pragma unroll
    for (int ks = 0; ks < 8; ++ks) {
      c = __builtin_amdgcn_wmma_f32_16x16x32_f16(false, A[ks], false, B[ks],
                                                 (short)0, c, false, false);
    }

    const int col = nt * 16 + lm;                // global centroid index
#pragma unroll
    for (int j = 0; j < 8; ++j) {
      const int row = rowbase + j + hi * 8;      // C/D: lanes 0-15 -> M=j, 16-31 -> M=j+8
      const int spk = row / U_UTT;
      const float s = c[j];
      if (col == spk) pos[j] = s;                // positive term, excluded from neg sum
      else            acc[j] += __expf(s - 1.0f); // |sim|<=1 -> fixed LSE shift
    }

    __syncthreads();                             // all readers done with bsh[cur]
    if (widx == 0 && nt + 1 < S_SPK / 16) {
      wait_async_all();                          // tile nt+1 landed in LDS
    }
    __syncthreads();                             // publish bsh[cur^1]
  }

  // reduce over the 16 N-lanes of each half-wave
#pragma unroll
  for (int j = 0; j < 8; ++j) {
    float a = acc[j], p = pos[j];
#pragma unroll
    for (int m = 8; m >= 1; m >>= 1) {
      a += __shfl_xor(a, m, 16);
      p += __shfl_xor(p, m, 16);
    }
    acc[j] = a; pos[j] = p;
  }
  if (lm == 0) {
#pragma unroll
    for (int j = 0; j < 8; ++j) {
      const int row = rowbase + j + hi * 8;
      loss[row] = -pos[j] + 1.0f + __logf(acc[j]);
    }
  }
}

// ---------------- phase 3: deterministic mean ------------------------------
__global__ void __launch_bounds__(256) final_red(const float* __restrict__ loss,
                                                 float* __restrict__ out) {
  __shared__ float sm[256];
  float s = 0.f;
  for (int i = threadIdx.x; i < B_ROWS; i += 256) s += loss[i];
  sm[threadIdx.x] = s;
  __syncthreads();
  for (int stride = 128; stride > 0; stride >>= 1) {
    if ((int)threadIdx.x < stride) sm[threadIdx.x] += sm[threadIdx.x + stride];
    __syncthreads();
  }
  if (threadIdx.x == 0) out[0] = sm[0] / (float)B_ROWS;
}

extern "C" void kernel_launch(void* const* d_in, const int* in_sizes, int n_in,
                              void* d_out, int out_size, void* d_ws, size_t ws_size,
                              hipStream_t stream) {
  const float* E = (const float*)d_in[0];
  // labels (d_in[1]) encode repeat(arange(S),U): speaker(row) = row / U, used inline.
  (void)in_sizes; (void)n_in; (void)out_size; (void)ws_size;

  char* ws = (char*)d_ws;
  _Float16* EnH = (_Float16*)ws;                                     // 10.0 MiB
  _Float16* CnH = (_Float16*)(ws + (size_t)B_ROWS * D_DIM * 2);      // +1.0 MiB (32B aligned)
  float*    lossbuf = (float*)(ws + (size_t)B_ROWS * D_DIM * 2
                                  + (size_t)S_SPK * D_DIM * 2);      // +80 KiB

  hipLaunchKernelGGL(prep_rows, dim3(B_ROWS / 8), dim3(256), 0, stream, E, EnH);
  hipLaunchKernelGGL(prep_cent, dim3(S_SPK / 8), dim3(256), 0, stream, E, CnH);
  hipLaunchKernelGGL(ge2e_sim, dim3(B_ROWS / 16 / 8), dim3(256), 0, stream,
                     EnH, CnH, lossbuf);
  hipLaunchKernelGGL(final_red, dim3(1), dim3(256), 0, stream, lossbuf, (float*)d_out);
}
